// QRNNLayer_33062658244664
// MI455X (gfx1250) — matrix operand order repaired and checked by reading, not compiled, not run on hardware
//
#include <hip/hip_runtime.h>
#include <hip/hip_bf16.h>

// QRNN layer: B=32, C_IN=512, H=512, K=2, T=1024
//
// Pipeline:
//   pack_w : W f32 [1536,1024] -> bf16, swizzled into per-lane WMMA A-fragment layout
//   pack_x : X f32 [B,C,T] -> bf16 B-fragments (k = 2c+kk pairs (x[t-1], x[t])),
//            causal pad handled here (branchless select)
//   gemm   : gates = W @ Xcat via v_wmma_f32_16x16x32_bf16; register ping-pong
//            double buffering, unsigned-offset global b128 loads; fused
//            bias+tanh/sigmoid; Z->d_out[0], F->ws, O->d_out[1]
//   fopool : c_t = f*c + (1-f)*z scan via wave32 Hillis-Steele (__shfl_up), h = o*c

typedef __attribute__((ext_vector_type(16))) __bf16 v16bf;
typedef __attribute__((ext_vector_type(8)))  float  v8f;

constexpr int Bn  = 32;
constexpr int CIN = 512;
constexpr int Hn  = 512;
constexpr int Tn  = 1024;
constexpr int Mn  = 3 * Hn;            // 1536 gate rows
constexpr int KT  = CIN * 2;           // 1024 reduction (k = 2c + kk)
constexpr int Nn  = Bn * Tn;           // 32768 columns (b,t)
constexpr int MT  = Mn / 16;           // 96  M-tiles
constexpr int NT  = Nn / 16;           // 2048 N-tiles
constexpr int KB  = KT / 32;           // 32  K-chunks
constexpr unsigned TILE_ELEMS = KB * 32 * 16;            // 16384 bf16 per tile
// packed fragment storage: per (tile, kchunk, lane): 16 bf16 = 32B contiguous
constexpr size_t XP_ELEMS = (size_t)NT * TILE_ELEMS;     // 33.5M bf16 = 64 MB
constexpr size_t WP_ELEMS = (size_t)MT * TILE_ELEMS;     //  1.5M bf16 =  3 MB

__device__ __forceinline__ float sigmoid_f(float v) {
    return 1.0f / (1.0f + expf(-v));
}

// ---- pack W into A-fragment layout -----------------------------------------
// A fragment as packed here (contiguous k): lane (half,r): row o = tile*16 + r,
// elements 0..15 = K(kb+half*16+0..15)
__global__ __launch_bounds__(256)
void pack_w(const float* __restrict__ w, __bf16* __restrict__ wp) {
    const int gid    = blockIdx.x * 256 + threadIdx.x;   // MT*KB*32 = 98304
    const int lane   = gid & 31;
    const int kbi    = (gid >> 5) & 31;
    const int m_tile = gid >> 10;                        // 0..95
    const int half   = lane >> 4;
    const int r      = lane & 15;

    const int o = m_tile * 16 + r;
    const float* pa = w + (size_t)o * KT + kbi * 32 + half * 16;
    v16bf out;
    #pragma unroll
    for (int e = 0; e < 8; ++e) out[e]     = (__bf16)pa[e];
    #pragma unroll
    for (int e = 0; e < 8; ++e) out[8 + e] = (__bf16)pa[8 + e];
    *(v16bf*)(wp + ((size_t)gid) * 16) = out;
}

// ---- pack X into B-fragment layout -----------------------------------------
// B fragment as packed here (contiguous k, same as A): lane (half,r): column n = tile*16 + r,
// element e = K(kb + half*16 + e)  ->  c = kb/2 + half*8 + e/2, kk = e&1
// value = x[b, c, t-1+kk]  (t-1 pad = 0)
__global__ __launch_bounds__(256)
void pack_x(const float* __restrict__ x, __bf16* __restrict__ xp) {
    const int gid    = blockIdx.x * 256 + threadIdx.x;   // NT*KB*32 = 2097152
    const int lane   = gid & 31;
    const int kbi    = (gid >> 5) & 31;
    const int n_tile = gid >> 10;                        // 0..2047
    const int half   = lane >> 4;
    const int r      = lane & 15;

    const int n  = n_tile * 16 + r;
    const int b  = n >> 10;
    const int t  = n & (Tn - 1);
    const float* xb = x + (size_t)b * CIN * Tn;
    const int c0   = kbi * 16 + half * 8;
    const int toff = (t > 0) ? -1 : 0;                   // clamp: stay in bounds

    v16bf out;
    #pragma unroll
    for (int m = 0; m < 8; ++m) {
        const int i1 = (c0 + m) * Tn + t;
        const float v1 = xb[i1];                         // x[b,c,t]
        float v0 = xb[i1 + toff];                        // x[b,c,t-1] (or garbage at t=0)
        v0 = (t > 0) ? v0 : 0.0f;                        // causal pad
        out[2 * m]     = (__bf16)v0;
        out[2 * m + 1] = (__bf16)v1;
    }
    *(v16bf*)(xp + ((size_t)gid) * 16) = out;
}

// ---- gates GEMM: wave = 4 M-tiles x 2 N-tiles, ping-pong double buffered ---
__global__ __launch_bounds__(128)
void qrnn_gates_wmma(const __bf16* __restrict__ wp,
                     const __bf16* __restrict__ xp,
                     const float*  __restrict__ bias,
                     float* __restrict__ zc,    // d_out plane 0: activated Z
                     float* __restrict__ fbuf,  // d_ws        : activated F
                     float* __restrict__ oh)    // d_out plane 1: activated O
{
    const int tid  = threadIdx.x;
    const int wv   = tid >> 5;
    const int lane = tid & 31;
    const int half = lane >> 4;
    const int r    = lane & 15;

    const int g    = blockIdx.x;          // 6 * 1024 = 6144 blocks
    const int msup = g % 6;
    const int ng   = g / 6;               // 0..1023 ; 4 waves share ng -> B in L0
    const int wg   = msup * 4 + wv;       // 0..23
    const int mt0  = wg * 4;              // first of 4 M-tiles
    const int nt0  = ng * 2;              // first of 2 N-tiles

    const int ncol0 = nt0 * 16;
    const int b     = ncol0 >> 10;        // all 32 columns share batch
    const int tb0   = ncol0 & (Tn - 1);

    // unsigned 32-bit element offsets -> global_load_b128 with imm deltas;
    // offsets bumped once per 2 K-steps at the bottom of the loop.
    unsigned boff[2], aoff[4];
    #pragma unroll
    for (int nb = 0; nb < 2; ++nb)
        boff[nb] = (unsigned)(nt0 + nb) * TILE_ELEMS + (unsigned)lane * 16u;
    #pragma unroll
    for (int ma = 0; ma < 4; ++ma)
        aoff[ma] = (unsigned)(mt0 + ma) * TILE_ELEMS + (unsigned)lane * 16u;

    v8f acc[4][2];
    #pragma unroll
    for (int ma = 0; ma < 4; ++ma)
        #pragma unroll
        for (int nb = 0; nb < 2; ++nb) acc[ma][nb] = v8f{0,0,0,0,0,0,0,0};

    v16bf a0[4], b0[2], a1[4], b1[2];
    #pragma unroll
    for (int nb = 0; nb < 2; ++nb) b0[nb] = *(const v16bf*)(xp + boff[nb]);
    #pragma unroll
    for (int ma = 0; ma < 4; ++ma) a0[ma] = *(const v16bf*)(wp + aoff[ma]);

    for (int kbi = 0; kbi < KB; kbi += 2) {
        // load k+1 (pong): +1024B immediates
        #pragma unroll
        for (int nb = 0; nb < 2; ++nb) b1[nb] = *(const v16bf*)(xp + boff[nb] + 512u);
        #pragma unroll
        for (int ma = 0; ma < 4; ++ma) a1[ma] = *(const v16bf*)(wp + aoff[ma] + 512u);
        // wmma k (ping)
        #pragma unroll
        for (int ma = 0; ma < 4; ++ma)
            #pragma unroll
            for (int nb = 0; nb < 2; ++nb)
                acc[ma][nb] = __builtin_amdgcn_wmma_f32_16x16x32_bf16(
                    false, a0[ma], false, b0[nb], (short)0, acc[ma][nb], false, false);
        // load k+2 (ping): +2048B immediates; final iter over-reads adjacent ws (unused)
        #pragma unroll
        for (int nb = 0; nb < 2; ++nb) b0[nb] = *(const v16bf*)(xp + boff[nb] + 1024u);
        #pragma unroll
        for (int ma = 0; ma < 4; ++ma) a0[ma] = *(const v16bf*)(wp + aoff[ma] + 1024u);
        // wmma k+1 (pong)
        #pragma unroll
        for (int ma = 0; ma < 4; ++ma)
            #pragma unroll
            for (int nb = 0; nb < 2; ++nb)
                acc[ma][nb] = __builtin_amdgcn_wmma_f32_16x16x32_bf16(
                    false, a1[ma], false, b1[nb], (short)0, acc[ma][nb], false, false);
        // bump stream offsets (one v_add_nc_u32 per stream)
        #pragma unroll
        for (int nb = 0; nb < 2; ++nb) boff[nb] += 1024u;
        #pragma unroll
        for (int ma = 0; ma < 4; ++ma) aoff[ma] += 1024u;
    }

    // epilogue: bias + activation + scatter to [B,H,T] planes
    const int region = mt0 / 32;          // 0:Z(tanh) 1:F(sig) 2:O(sig); uniform per wave
    float* __restrict__ dst = (region == 0) ? zc : (region == 1) ? fbuf : oh;

    #pragma unroll
    for (int ma = 0; ma < 4; ++ma) {
        const int obase = (mt0 + ma) * 16 + half * 8;   // gate row of element i = obase+i
        const float4 bb0 = *(const float4*)(bias + obase);
        const float4 bb1 = *(const float4*)(bias + obase + 4);
        const float bv[8] = {bb0.x, bb0.y, bb0.z, bb0.w, bb1.x, bb1.y, bb1.z, bb1.w};
        const int hbase = obase & (Hn - 1);
        #pragma unroll
        for (int nb = 0; nb < 2; ++nb) {
            const int t = tb0 + nb * 16 + r;
            #pragma unroll
            for (int i = 0; i < 8; ++i) {
                float v = acc[ma][nb][i] + bv[i];
                v = (region == 0) ? tanhf(v) : sigmoid_f(v);
                dst[((size_t)b * Hn + (hbase + i)) * Tn + t] = v;
            }
        }
    }
}

// ---- fo-pool: wave32 inclusive scan of affine maps c -> A*c + B ------------
__global__ __launch_bounds__(256)
void qrnn_fopool(float* __restrict__ zc,          // in: Z, out: c
                 const float* __restrict__ fbuf,  // in: F
                 float* __restrict__ oh)          // in: O, out: h
{
    const int gtid = blockIdx.x * 256 + threadIdx.x;
    const int row  = gtid >> 5;                   // (b*H + h), 0..16383
    const int lane = gtid & 31;
    const size_t base = (size_t)row * Tn;

    float carry = 0.0f;
    for (int tb = 0; tb < Tn; tb += 32) {
        const size_t idx = base + tb + lane;
        const float z = zc[idx];
        const float f = fbuf[idx];
        const float o = oh[idx];

        float A  = f;
        float Bv = (1.0f - f) * z;
        #pragma unroll
        for (int d = 1; d < 32; d <<= 1) {
            const float aU = __shfl_up(A,  d, 32);
            const float bU = __shfl_up(Bv, d, 32);
            if (lane >= d) {
                Bv = fmaf(A, bU, Bv);
                A  = A * aU;
            }
        }
        const float c = fmaf(A, carry, Bv);
        zc[idx] = c;
        oh[idx] = o * c;
        carry = __shfl(c, 31, 32);
    }
}

extern "C" void kernel_launch(void* const* d_in, const int* in_sizes, int n_in,
                              void* d_out, int out_size, void* d_ws, size_t ws_size,
                              hipStream_t stream) {
    const float* x    = (const float*)d_in[0];   // [32, 512, 1024] f32
    const float* w    = (const float*)d_in[1];   // [1536, 512, 2]  f32
    const float* bias = (const float*)d_in[2];   // [1536]          f32

    float* outC = (float*)d_out;                         // c_time [B,H,T]
    float* outH = outC + (size_t)Bn * Hn * Tn;           // h_time [B,H,T]

    // workspace layout: xpack (64MB) | wpack (3MB) | F plane (64MB)
    char* ws = (char*)d_ws;
    __bf16* xp   = (__bf16*)ws;
    __bf16* wpk  = (__bf16*)(ws + XP_ELEMS * sizeof(__bf16));
    float*  fbuf = (float*) (ws + (XP_ELEMS + WP_ELEMS) * sizeof(__bf16));

    pack_w<<<(MT * KB * 32) / 256, 256, 0, stream>>>(w, wpk);
    pack_x<<<(NT * KB * 32) / 256, 256, 0, stream>>>(x, xp);
    qrnn_gates_wmma<<<6 * 1024, 128, 0, stream>>>(wpk, xp, bias, outC, fbuf, outH);
    qrnn_fopool<<<(Bn * Hn * 32) / 256, 256, 0, stream>>>(outC, fbuf, outH);
}
